// GGNN_72232759984373
// MI455X (gfx1250) — matrix operand order repaired and checked by reading, not compiled, not run on hardware
//
#include <hip/hip_runtime.h>

// ---------------------------------------------------------------------------
// GGNN forward for MI455X (gfx1250), wave32, bf16 WMMA with f32 accumulation.
// v3: fix B-fragment K layout per ISA (lanes 0-15: K=0-7,16-23, lanes 16-31:
//     K=8-15,24-31, 2 values per VGPR); otherwise identical tiling to v2.
// ---------------------------------------------------------------------------

#define N_NODES 8192
#define N_EDGES 131072
#define N_TOT   (N_NODES + N_EDGES)   // 139264
#define HDIM    128

typedef __attribute__((ext_vector_type(16))) __bf16 bf16x16;
typedef __attribute__((ext_vector_type(8)))  __bf16 bf16x8;
typedef __attribute__((ext_vector_type(8)))  float  v8f;

__device__ __forceinline__ __bf16 f2bf(float f) { return (__bf16)f; }

// ---- WMMA fragment loaders -------------------------------------------------
// A (16x32 bf16), ISA table: lane L holds row M=L%16; with h=L/16:
//   VGPR j (j=0..3): K = 8h+2j, 8h+2j+1   -> a[0..7]  = A[M][t + 8h + j]
//   VGPR j (j=4..7): K = 16+8h+...        -> a[8..15] = A[M][t + 16 + 8h + j]
__device__ __forceinline__ bf16x16 load_a_frag(const __bf16* As, int lda,
                                               int l16, int t, int half) {
  const __bf16* p = As + l16 * lda + t + half * 8;
  bf16x8 lo = *(const bf16x8*)p;
  bf16x8 hi = *(const bf16x8*)(p + 16);
  bf16x16 a;
#pragma unroll
  for (int j = 0; j < 8; ++j) { a[j] = lo[j]; a[8 + j] = hi[j]; }
  return a;
}

// B (32x16 bf16), same K order as A: with h=L/16:
//   b[0..7]  = B[t + 8h + j][n],  b[8..15] = B[t + 16 + 8h + j][n]  (packed [N][K] weight)
__device__ __forceinline__ bf16x16 load_b_packed(const __bf16* Bn, int t, int half) {
  const __bf16* p = Bn + t + half * 8;
  bf16x8 lo = *(const bf16x8*)p;
  bf16x8 hi = *(const bf16x8*)(p + 16);
  bf16x16 b;
#pragma unroll
  for (int j = 0; j < 8; ++j) { b[j] = lo[j]; b[8 + j] = hi[j]; }
  return b;
}

__device__ __forceinline__ v8f wmma_bf16(bf16x16 a, bf16x16 b, v8f c) {
  return __builtin_amdgcn_wmma_f32_16x16x32_bf16(false, a, false, b,
                                                 (short)0, c, false, false);
}

// ---------------------------------------------------------------------------
// Weight pack: out[n*K + k] = trans ? W[k*N + n] : W[n*K + k], converted bf16.
// ---------------------------------------------------------------------------
__global__ void pack_weights(const float* __restrict__ W, __bf16* __restrict__ out,
                             int K, int N, int trans) {
  const int i = blockIdx.x * blockDim.x + threadIdx.x;
  if (i >= N * K) return;
  const int n = i / K, k = i - n * K;
  const float v = trans ? W[(size_t)k * N + n] : W[i];
  out[i] = f2bf(v);
}

// ---------------------------------------------------------------------------
// GEMM:  C[M x N] = act(A[M x K] @ Bp^T + bias),  Bp is bf16 [N][K].
// 128 threads = 4 waves; workgroup owns 64 rows (4 M-tiles); wave w owns
// N-tile (blockIdx.y*64 + 16w); B fragment reused across 4 M-tiles.
// grid = (M/64, N/64).  K in {64,128}.  act: 0=none, 1=relu.
// ---------------------------------------------------------------------------
__global__ __launch_bounds__(128) void gemm_bf16_v2(
    const float* __restrict__ A, const __bf16* __restrict__ Bp,
    const float* __restrict__ bias, float* __restrict__ C,
    int K, int N, int act) {
  __shared__ __align__(16) __bf16 As[64][136];
  const int tid  = threadIdx.x;
  const int row0 = blockIdx.x * 64;

  const int kq = K >> 2;  // K/4
  for (int i = tid; i < 64 * kq; i += 128) {
    const int r = i / kq, c4 = i - r * kq;
    const float4 v = ((const float4*)(A + (size_t)(row0 + r) * K))[c4];
    __bf16* d = &As[r][c4 * 4];
    d[0] = f2bf(v.x); d[1] = f2bf(v.y); d[2] = f2bf(v.z); d[3] = f2bf(v.w);
  }
  __syncthreads();

  const int wave = tid >> 5, lane = tid & 31;
  const int half = lane >> 4, l16 = lane & 15;
  const int n = blockIdx.y * 64 + wave * 16 + l16;
  const __bf16* Bn = Bp + (size_t)n * K;

  v8f acc0 = {}, acc1 = {}, acc2 = {}, acc3 = {};
  for (int t = 0; t < K; t += 32) {
    const bf16x16 b = load_b_packed(Bn, t, half);
    acc0 = wmma_bf16(load_a_frag(&As[0][0],  136, l16, t, half), b, acc0);
    acc1 = wmma_bf16(load_a_frag(&As[16][0], 136, l16, t, half), b, acc1);
    acc2 = wmma_bf16(load_a_frag(&As[32][0], 136, l16, t, half), b, acc2);
    acc3 = wmma_bf16(load_a_frag(&As[48][0], 136, l16, t, half), b, acc3);
  }

  const float bv = bias ? bias[n] : 0.0f;
  v8f accs[4] = {acc0, acc1, acc2, acc3};
#pragma unroll
  for (int mt = 0; mt < 4; ++mt) {
#pragma unroll
    for (int i = 0; i < 8; ++i) {
      float v = accs[mt][i] + bv;
      if (act) v = fmaxf(v, 0.0f);
      C[(size_t)(row0 + mt * 16 + i + 8 * half) * N + n] = v;
    }
  }
}

// ---------------------------------------------------------------------------
// Fused GRU: x <- GRU(agg, x), 32 rows per block (256 thr = 8 waves; wave w
// owns cols [16w,16w+16) over 2 M-tiles).  Packed bf16 wi/wh ([384][128]).
// ---------------------------------------------------------------------------
struct v8f2 { v8f a, b; };

__device__ __forceinline__ v8f2 gru_tile2(const __bf16* As, const __bf16* Wp,
                                          int wrow, int l16, int half) {
  const __bf16* Bn = Wp + (size_t)wrow * 128;
  v8f a0 = {}, a1 = {};
#pragma unroll
  for (int t = 0; t < 128; t += 32) {
    const bf16x16 b = load_b_packed(Bn, t, half);
    a0 = wmma_bf16(load_a_frag(As,            136, l16, t, half), b, a0);
    a1 = wmma_bf16(load_a_frag(As + 16 * 136, 136, l16, t, half), b, a1);
  }
  return {a0, a1};
}

__global__ __launch_bounds__(256) void gru_fused(
    float* __restrict__ x, const float* __restrict__ agg,
    const __bf16* __restrict__ wi, const __bf16* __restrict__ wh,
    const float* __restrict__ bi, const float* __restrict__ bh) {
  __shared__ __align__(16) __bf16 Ag[32][136];
  __shared__ __align__(16) __bf16 Ax[32][136];
  const int tid  = threadIdx.x;  // 256
  const int row0 = blockIdx.x * 32;

  for (int i = tid; i < 32 * 32; i += 256) {   // 32 rows x 32 float4s
    const int r = i >> 5, c4 = i & 31;
    const float4 vg = ((const float4*)(agg + (size_t)(row0 + r) * 128))[c4];
    const float4 vx = ((const float4*)(x   + (size_t)(row0 + r) * 128))[c4];
    __bf16* dg = &Ag[r][c4 * 4];
    __bf16* dx = &Ax[r][c4 * 4];
    dg[0] = f2bf(vg.x); dg[1] = f2bf(vg.y); dg[2] = f2bf(vg.z); dg[3] = f2bf(vg.w);
    dx[0] = f2bf(vx.x); dx[1] = f2bf(vx.y); dx[2] = f2bf(vx.z); dx[3] = f2bf(vx.w);
  }
  __syncthreads();

  const int wave = tid >> 5, lane = tid & 31;
  const int half = lane >> 4, l16 = lane & 15;
  const int c = wave * 16 + l16;  // output column 0..127

  const v8f2 ir = gru_tile2(&Ag[0][0], wi, c,       l16, half);
  const v8f2 hr = gru_tile2(&Ax[0][0], wh, c,       l16, half);
  const v8f2 iz = gru_tile2(&Ag[0][0], wi, 128 + c, l16, half);
  const v8f2 hz = gru_tile2(&Ax[0][0], wh, 128 + c, l16, half);
  const v8f2 in_ = gru_tile2(&Ag[0][0], wi, 256 + c, l16, half);
  const v8f2 hn  = gru_tile2(&Ax[0][0], wh, 256 + c, l16, half);

  const float bir = bi[c],       bhr = bh[c];
  const float biz = bi[128 + c], bhz = bh[128 + c];
  const float bin = bi[256 + c], bhn = bh[256 + c];

#pragma unroll
  for (int mt = 0; mt < 2; ++mt) {
    const v8f vir = mt ? ir.b : ir.a, vhr = mt ? hr.b : hr.a;
    const v8f viz = mt ? iz.b : iz.a, vhz = mt ? hz.b : hz.a;
    const v8f vin = mt ? in_.b : in_.a, vhn = mt ? hn.b : hn.a;
#pragma unroll
    for (int i = 0; i < 8; ++i) {
      const size_t idx = (size_t)(row0 + mt * 16 + i + 8 * half) * 128 + c;
      const float xo = x[idx];
      const float r  = 1.0f / (1.0f + expf(-(vir[i] + bir + vhr[i] + bhr)));
      const float z  = 1.0f / (1.0f + expf(-(viz[i] + biz + vhz[i] + bhz)));
      const float nn = tanhf(vin[i] + bin + r * (vhn[i] + bhn));
      x[idx] = (1.0f - z) * nn + z * xo;
    }
  }
}

// ---------------------------------------------------------------------------
// Sparse scatters & misc elementwise kernels
// ---------------------------------------------------------------------------
__global__ void init_x_kernel(float* __restrict__ x, const float* __restrict__ jvals) {
  const long long tid = (long long)blockIdx.x * blockDim.x + threadIdx.x;
  const int r = (int)(tid >> 5), q = (int)(tid & 31);
  if (r >= N_TOT) return;
  float4 v = {0.f, 0.f, 0.f, 0.f};
  if (r >= N_NODES && q == 0) { v.x = 1.0f; v.y = jvals[r - N_NODES]; }
  ((float4*)(x + (size_t)r * 128))[q] = v;
}

__global__ void zero_f32(float* __restrict__ p, int n) {
  const int i = blockIdx.x * blockDim.x + threadIdx.x;
  if (i < n) p[i] = 0.0f;
}

// agg[n_nodes+e] = m[row[e]] + m[col[e]]    (deterministic, no atomics)
__global__ void scatter_fac(const float* __restrict__ m, const int* __restrict__ row,
                            const int* __restrict__ col, float* __restrict__ agg) {
  const long long tid = (long long)blockIdx.x * blockDim.x + threadIdx.x;
  const int e = (int)(tid >> 5), q = (int)(tid & 31);
  if (e >= N_EDGES) return;
  const int r = row[e], cc = col[e];
  float4 a = ((const float4*)(m + (size_t)r  * 128))[q];
  float4 b = ((const float4*)(m + (size_t)cc * 128))[q];
  float4 o = {a.x + b.x, a.y + b.y, a.z + b.z, a.w + b.w};
  ((float4*)(agg + (size_t)(N_NODES + e) * 128))[q] = o;
}

// agg[row[e]] += m[fac_e];  agg[col[e]] += m[fac_e]
__global__ void scatter_node(const float* __restrict__ m, const int* __restrict__ row,
                             const int* __restrict__ col, float* __restrict__ agg) {
  const long long tid = (long long)blockIdx.x * blockDim.x + threadIdx.x;
  const int e = (int)(tid >> 7), f = (int)(tid & 127);
  if (e >= N_EDGES) return;
  const float v = m[(size_t)(N_NODES + e) * 128 + f];
  atomicAdd(&agg[(size_t)row[e] * 128 + f], v);
  atomicAdd(&agg[(size_t)col[e] * 128 + f], v);
}

// node_msgs[row[e]] += msgs[e]
__global__ void scatter_msgs(const float* __restrict__ msgs, const int* __restrict__ row,
                             float* __restrict__ node_msgs) {
  const long long tid = (long long)blockIdx.x * blockDim.x + threadIdx.x;
  const int e = (int)(tid >> 6), f = (int)(tid & 63);
  if (e >= N_EDGES) return;
  atomicAdd(&node_msgs[(size_t)row[e] * 64 + f], msgs[(size_t)e * 64 + f]);
}

// logits = hr2 @ ro_W3 + b3 ; softmax  (tiny: 8192 x 2, K=128)
__global__ void readout_final(const float* __restrict__ hr2, const float* __restrict__ W3,
                              const float* __restrict__ b3, float* __restrict__ out) {
  const int nidx = blockIdx.x * blockDim.x + threadIdx.x;
  if (nidx >= N_NODES) return;
  const float* h = hr2 + (size_t)nidx * 128;
  float l0 = b3[0], l1 = b3[1];
#pragma unroll 4
  for (int k = 0; k < 128; ++k) {
    const float hv = h[k];
    l0 += hv * W3[k * 2 + 0];
    l1 += hv * W3[k * 2 + 1];
  }
  const float mx = fmaxf(l0, l1);
  const float e0 = expf(l0 - mx), e1 = expf(l1 - mx);
  const float s = e0 + e1;
  out[nidx * 2 + 0] = e0 / s;
  out[nidx * 2 + 1] = e1 / s;
}

// ---------------------------------------------------------------------------
extern "C" void kernel_launch(void* const* d_in, const int* in_sizes, int n_in,
                              void* d_out, int out_size, void* d_ws, size_t ws_size,
                              hipStream_t stream) {
  const float* jvals  = (const float*)d_in[0];
  const int*   row    = (const int*)d_in[2];
  const int*   col    = (const int*)d_in[3];
  const float* conv_W = (const float*)d_in[4];
  const float* gru_wi = (const float*)d_in[5];
  const float* gru_wh = (const float*)d_in[6];
  const float* gru_bi = (const float*)d_in[7];
  const float* gru_bh = (const float*)d_in[8];
  const float* mp_W1  = (const float*)d_in[9];
  const float* mp_b1  = (const float*)d_in[10];
  const float* mp_W2  = (const float*)d_in[11];
  const float* mp_b2  = (const float*)d_in[12];
  const float* mp_W3  = (const float*)d_in[13];
  const float* mp_b3  = (const float*)d_in[14];
  const float* ro_W1  = (const float*)d_in[15];
  const float* ro_b1  = (const float*)d_in[16];
  const float* ro_W2  = (const float*)d_in[17];
  const float* ro_b2  = (const float*)d_in[18];
  const float* ro_W3  = (const float*)d_in[19];
  const float* ro_b3  = (const float*)d_in[20];

  // Workspace: three n_tot x 128 f32 buffers + packed bf16 weights.
  const size_t XB = (size_t)N_TOT * 128 * sizeof(float);   // 71,303,168 B
  float* x   = (float*)d_ws;
  float* m   = (float*)((char*)d_ws + XB);
  float* agg = (float*)((char*)d_ws + 2 * XB);
  float* hm1       = m;                                    // 131072 x 128
  float* hm2       = agg;                                  // 131072 x 128
  float* msgs      = m;                                    // 131072 x 64
  float* node_msgs = agg;                                  // 8192 x 64
  float* hr1       = m;                                    // 8192 x 128
  float* hr2       = (float*)((char*)agg + (8u << 20));    // 8192 x 128

  __bf16* wp = (__bf16*)((char*)d_ws + 3 * XB);
  __bf16* conv0_p = wp;                  // [128][128]
  __bf16* conv1_p = wp + 16384;          // [128][128]
  __bf16* wi_p    = wp + 32768;          // [384][128]
  __bf16* wh_p    = wp + 81920;          // [384][128]
  __bf16* mpW1_p  = wp + 131072;         // [128][128]
  __bf16* mpW2_p  = wp + 147456;         // [128][128]
  __bf16* mpW3_p  = wp + 163840;         // [64][128]
  __bf16* roW1_p  = wp + 172032;         // [128][64]
  __bf16* roW2_p  = wp + 180224;         // [128][128]

  // ---- pack weights to bf16 [N][K] ----
  pack_weights<<<64, 256, 0, stream>>>(conv_W,          conv0_p, 128, 128, 1);
  pack_weights<<<64, 256, 0, stream>>>(conv_W + 16384,  conv1_p, 128, 128, 1);
  pack_weights<<<192, 256, 0, stream>>>(gru_wi,         wi_p,    128, 384, 0);
  pack_weights<<<192, 256, 0, stream>>>(gru_wh,         wh_p,    128, 384, 0);
  pack_weights<<<64, 256, 0, stream>>>(mp_W1,           mpW1_p,  128, 128, 1);
  pack_weights<<<64, 256, 0, stream>>>(mp_W2,           mpW2_p,  128, 128, 1);
  pack_weights<<<32, 256, 0, stream>>>(mp_W3,           mpW3_p,  128, 64,  1);
  pack_weights<<<32, 256, 0, stream>>>(ro_W1,           roW1_p,  64,  128, 1);
  pack_weights<<<64, 256, 0, stream>>>(ro_W2,           roW2_p,  128, 128, 1);

  // ---- init x ----
  init_x_kernel<<<(N_TOT * 32) / 256, 256, 0, stream>>>(x, jvals);

  // ---- 2 GGNN layers ----
  const __bf16* conv_p[2] = {conv0_p, conv1_p};
  for (int l = 0; l < 2; ++l) {
    gemm_bf16_v2<<<dim3(N_TOT / 64, 2), 128, 0, stream>>>(
        x, conv_p[l], nullptr, m, 128, 128, 0);
    zero_f32<<<(N_NODES * 128) / 256, 256, 0, stream>>>(agg, N_NODES * 128);
    scatter_fac<<<(N_EDGES * 32) / 256, 256, 0, stream>>>(m, row, col, agg);
    scatter_node<<<(N_EDGES * 128) / 256, 256, 0, stream>>>(m, row, col, agg);
    gru_fused<<<N_TOT / 32, 256, 0, stream>>>(x, agg, wi_p, wh_p, gru_bi, gru_bh);
  }

  // ---- edge MLP on factor rows ----
  const float* xf = x + (size_t)N_NODES * 128;
  gemm_bf16_v2<<<dim3(N_EDGES / 64, 2), 128, 0, stream>>>(xf,  mpW1_p, mp_b1, hm1, 128, 128, 1);
  gemm_bf16_v2<<<dim3(N_EDGES / 64, 2), 128, 0, stream>>>(hm1, mpW2_p, mp_b2, hm2, 128, 128, 1);
  gemm_bf16_v2<<<dim3(N_EDGES / 64, 1), 128, 0, stream>>>(hm2, mpW3_p, mp_b3, msgs, 128, 64, 0);

  // ---- aggregate messages per node ----
  zero_f32<<<(N_NODES * 64) / 256, 256, 0, stream>>>(node_msgs, N_NODES * 64);
  scatter_msgs<<<(N_EDGES * 64) / 256, 256, 0, stream>>>(msgs, row, node_msgs);

  // ---- readout ----
  gemm_bf16_v2<<<dim3(N_NODES / 64, 2), 128, 0, stream>>>(node_msgs, roW1_p, ro_b1, hr1, 64, 128, 1);
  gemm_bf16_v2<<<dim3(N_NODES / 64, 2), 128, 0, stream>>>(hr1,       roW2_p, ro_b2, hr2, 128, 128, 1);
  readout_final<<<N_NODES / 256, 256, 0, stream>>>(hr2, ro_W3, ro_b3, (float*)d_out);
}
